// MultiScaleLiquidRNN_80693845557613
// MI455X (gfx1250) — matrix operand order repaired and checked by reading, not compiled, not run on hardware
//
#include <hip/hip_runtime.h>
#include <math.h>

// ---------------------------------------------------------------------------
// MultiScaleLiquidRNN on gfx1250 (MI455X), f16 WMMA + f32 accumulate.
// SEQ=256, B=64, IN=512, H=1024, OUT=256, L=2, DT=0.05
// ---------------------------------------------------------------------------

typedef _Float16 half_t;
typedef __attribute__((ext_vector_type(16))) _Float16 v16h;
typedef __attribute__((ext_vector_type(8)))  _Float16 v8h;
typedef __attribute__((ext_vector_type(8)))  float    v8f;

#define SEQ_N   256
#define BATCH_N 64
#define IN_N    512
#define HID_N   1024
#define OUT_N   256
#define DT_C    0.05f

// ---------------------------------------------------------------------------
// CDNA5 16-bit operand layouts (ISA 7.12.2, V_WMMA_F32_16X16X32_F16):
//
// A (16x32, row-major source [M,K]):
//   lane l: row M = l&15 ; holds K = koff..koff+7 and K = koff+16..koff+23
//   with koff = (l>>4)*8  -> two discontiguous 16-byte chunks.
//
// B (32x16; source W row-major [N,K], B = W^T):
//   extrapolating the
//   documented sparse 16-bit B layout (Vn+0..7: lanes 0-15 hold K=0-15,
//   lanes 16-31 hold K=16-31) to dense K=32:
//   lane l: col N = l&15 ; holds K = koff..koff+7 and K = koff+16..koff+23
//   with koff = (l>>4)*8 (same K order as A).
//
// C/D (f32, 8 VGPRs): lane l -> N = l&15, elem r -> M = r + 8*(l>>4)
// ---------------------------------------------------------------------------
__device__ __forceinline__ v16h load_fragA(const half_t* __restrict__ M,
                                           int row0, int ld, int k0, int lane) {
    int r    = row0 + (lane & 15);
    int koff = (lane >> 4) * 8;
    const half_t* p = M + (size_t)r * ld + k0 + koff;
    union { v16h v; v8h h[2]; } u;
    u.h[0] = *(const v8h*)(p);        // K = koff .. koff+7
    u.h[1] = *(const v8h*)(p + 16);   // K = koff+16 .. koff+23
    return u.v;
}

__device__ __forceinline__ v16h load_fragB(const half_t* __restrict__ W,
                                           int col0, int ld, int k0, int lane) {
    int n    = col0 + (lane & 15);
    int koff = (lane >> 4) * 8;
    const half_t* p = W + (size_t)n * ld + k0 + koff;
    union { v16h v; v8h h[2]; } u;
    u.h[0] = *(const v8h*)(p);        // K = koff .. koff+7
    u.h[1] = *(const v8h*)(p + 16);   // K = koff+16 .. koff+23
    return u.v;
}

__device__ __forceinline__ v8f wmma16(v16h a, v16h b, v8f c) {
    return __builtin_amdgcn_wmma_f32_16x16x32_f16(
        /*neg_a=*/false, a, /*neg_b=*/false, b,
        /*c_mod=*/(short)0, c, /*reuse_a=*/false, /*reuse_b=*/false);
}

// ---------------------------------------------------------------------------
// f32 -> f16 conversion
// ---------------------------------------------------------------------------
__global__ void cvt_f16_kernel(const float* __restrict__ src,
                               half_t* __restrict__ dst, int n) {
    int i = blockIdx.x * blockDim.x + threadIdx.x;
    if (i < n) dst[i] = (half_t)src[i];
}

// ---------------------------------------------------------------------------
// Generic GEMM: D[M,N] = act(A[M,K] @ W[N,K]^T + bias)
// Each wave: two adjacent 16x16 N-tiles (independent accumulators, shared A).
// ACT: 0 = none, 1 = relu.  OUT16: store f16, else f32.
// ---------------------------------------------------------------------------
template<int ACT, bool OUT16>
__global__ void gemm_bias_kernel(const half_t* __restrict__ A, int lda,
                                 const half_t* __restrict__ W, int ldw,
                                 const float* __restrict__ bias,
                                 half_t* __restrict__ out16,
                                 float*  __restrict__ out32, int ldo,
                                 int ntilesN, int K) {
    const int lane   = threadIdx.x & 31;
    const int pair   = blockIdx.x * 8 + (threadIdx.x >> 5);
    const int npairs = ntilesN >> 1;
    const int tm     = pair / npairs;
    const int tn0    = (pair % npairs) * 2;

    v8f acc[2] = {{}, {}};
    for (int k0 = 0; k0 < K; k0 += 32) {
        v16h a  = load_fragA(A, tm * 16, lda, k0, lane);
        v16h b0 = load_fragB(W, tn0 * 16,      ldw, k0, lane);
        v16h b1 = load_fragB(W, tn0 * 16 + 16, ldw, k0, lane);
        acc[0] = wmma16(a, b0, acc[0]);
        acc[1] = wmma16(a, b1, acc[1]);
    }

    const int m0 = tm * 16 + (lane >> 4) * 8;
    #pragma unroll
    for (int t = 0; t < 2; ++t) {
        const int n  = (tn0 + t) * 16 + (lane & 15);
        const float bv = bias[n];
        #pragma unroll
        for (int r = 0; r < 8; ++r) {
            float v = acc[t][r] + bv;
            if (ACT == 1) v = fmaxf(v, 0.0f);
            if (OUT16) out16[(size_t)(m0 + r) * ldo + n] = (half_t)v;
            else       out32[(size_t)(m0 + r) * ldo + n] = v;
        }
    }
}

// ---------------------------------------------------------------------------
// Liquid cell step for one layer:
//   h_new = h + (DT/(tau+1e-3)) * tanh(inp @ Wcx^T + h @ Wch^T + b_c)
// Wcx = Wc[:, 0:H], Wch = Wc[:, H:2H]  (row-major [H, 2H], shared layers).
// h_f32 updated in place (element-private: read-then-write by owning lane);
// f16 shadow double-buffered by the caller (cross-block operand).
// ---------------------------------------------------------------------------
__global__ void liquid_step_kernel(const half_t* __restrict__ inp,    // [B,H] f16
                                   const half_t* __restrict__ h_in16, // [B,H] f16
                                   float*  __restrict__ h_f32,        // [B,H] in/out
                                   half_t* __restrict__ h_out16,      // [B,H] f16
                                   const half_t* __restrict__ Wc,     // [H,2H] f16
                                   const float* __restrict__ b_c,
                                   const float* __restrict__ tau_p) {
    const int lane   = threadIdx.x & 31;
    const int pair   = blockIdx.x * 8 + (threadIdx.x >> 5);
    const int npairs = (HID_N / 16) >> 1;           // 32
    const int tm     = pair / npairs;               // 0..3
    const int tn0    = (pair % npairs) * 2;

    const half_t* Wcx = Wc;
    const half_t* Wch = Wc + HID_N;                 // column offset H (16B aligned)

    v8f acc[2] = {{}, {}};
    for (int k0 = 0; k0 < HID_N; k0 += 32) {
        v16h a  = load_fragA(inp, tm * 16, HID_N, k0, lane);
        v16h b0 = load_fragB(Wcx, tn0 * 16,      2 * HID_N, k0, lane);
        v16h b1 = load_fragB(Wcx, tn0 * 16 + 16, 2 * HID_N, k0, lane);
        acc[0] = wmma16(a, b0, acc[0]);
        acc[1] = wmma16(a, b1, acc[1]);
    }
    for (int k0 = 0; k0 < HID_N; k0 += 32) {
        v16h a  = load_fragA(h_in16, tm * 16, HID_N, k0, lane);
        v16h b0 = load_fragB(Wch, tn0 * 16,      2 * HID_N, k0, lane);
        v16h b1 = load_fragB(Wch, tn0 * 16 + 16, 2 * HID_N, k0, lane);
        acc[0] = wmma16(a, b0, acc[0]);
        acc[1] = wmma16(a, b1, acc[1]);
    }

    const float scale = DT_C / (tau_p[0] + 0.001f);
    const int m0 = tm * 16 + (lane >> 4) * 8;
    #pragma unroll
    for (int t = 0; t < 2; ++t) {
        const int n  = (tn0 + t) * 16 + (lane & 15);
        const float bv = b_c[n];
        #pragma unroll
        for (int r = 0; r < 8; ++r) {
            size_t idx = (size_t)(m0 + r) * HID_N + n;
            float h  = h_f32[idx];
            float hn = h + scale * tanhf(acc[t][r] + bv);
            h_f32[idx]   = hn;
            h_out16[idx] = (half_t)hn;
        }
    }
}

// ---------------------------------------------------------------------------
// State init / final-hidden copy
// ---------------------------------------------------------------------------
__global__ void init_state_kernel(float* __restrict__ h0f, float* __restrict__ h1f,
                                  half_t* __restrict__ h016, half_t* __restrict__ h116) {
    int i = blockIdx.x * blockDim.x + threadIdx.x;   // 0 .. B*H-1
    h0f[i] = 0.0f; h1f[i] = 0.0f;
    h016[i] = (half_t)0.0f; h116[i] = (half_t)0.0f;
}

__global__ void copy_hidden_kernel(const float* __restrict__ h0,
                                   const float* __restrict__ h1,
                                   float* __restrict__ dst) {
    int i = blockIdx.x * blockDim.x + threadIdx.x;   // 0 .. B*H-1
    dst[i] = h0[i];
    dst[BATCH_N * HID_N + i] = h1[i];
}

// ---------------------------------------------------------------------------
// Host side
// ---------------------------------------------------------------------------
extern "C" void kernel_launch(void* const* d_in, const int* in_sizes, int n_in,
                              void* d_out, int out_size, void* d_ws, size_t ws_size,
                              hipStream_t stream) {
    const float* x    = (const float*)d_in[0];   // [S,B,IN]
    const float* W_in = (const float*)d_in[1];   // [H,IN]
    const float* b_in = (const float*)d_in[2];   // [H]
    const float* W_c  = (const float*)d_in[3];   // [H,2H]
    const float* b_c  = (const float*)d_in[4];   // [H]
    const float* taus = (const float*)d_in[5];   // [L]
    const float* W_o1 = (const float*)d_in[6];   // [H/2,H]
    const float* b_o1 = (const float*)d_in[7];   // [H/2]
    const float* W_o2 = (const float*)d_in[8];   // [OUT,H/2]
    const float* b_o2 = (const float*)d_in[9];   // [OUT]
    float* out = (float*)d_out;                  // [S,B,OUT] ++ [L,B,H]

    // ---- workspace layout (f16 operands + f32 state) ----
    char* w = (char*)d_ws;
    half_t* x16   = (half_t*)w; w += (size_t)SEQ_N * BATCH_N * IN_N * 2;
    half_t* Win16 = (half_t*)w; w += (size_t)HID_N * IN_N * 2;
    half_t* Wc16  = (half_t*)w; w += (size_t)HID_N * 2 * HID_N * 2;
    half_t* Wo116 = (half_t*)w; w += (size_t)(HID_N / 2) * HID_N * 2;
    half_t* Wo216 = (half_t*)w; w += (size_t)OUT_N * (HID_N / 2) * 2;
    half_t* xp16  = (half_t*)w; w += (size_t)SEQ_N * BATCH_N * HID_N * 2;
    half_t* h0_16[2]; h0_16[0] = (half_t*)w; w += (size_t)BATCH_N * HID_N * 2;
                      h0_16[1] = (half_t*)w; w += (size_t)BATCH_N * HID_N * 2;
    half_t* h1_16[2]; h1_16[0] = (half_t*)w; w += (size_t)BATCH_N * HID_N * 2;
                      h1_16[1] = (half_t*)w; w += (size_t)BATCH_N * HID_N * 2;
    float*  h0f   = (float*)w;  w += (size_t)BATCH_N * HID_N * 4;
    float*  h1f   = (float*)w;  w += (size_t)BATCH_N * HID_N * 4;
    half_t* o1_16 = (half_t*)w; w += (size_t)BATCH_N * (HID_N / 2) * 2;

    // ---- f32 -> f16 conversions ----
    auto cvt = [&](const float* s, half_t* d, int n) {
        cvt_f16_kernel<<<(n + 255) / 256, 256, 0, stream>>>(s, d, n);
    };
    cvt(x,    x16,   SEQ_N * BATCH_N * IN_N);
    cvt(W_in, Win16, HID_N * IN_N);
    cvt(W_c,  Wc16,  HID_N * 2 * HID_N);
    cvt(W_o1, Wo116, (HID_N / 2) * HID_N);
    cvt(W_o2, Wo216, OUT_N * (HID_N / 2));

    // ---- zero hidden state ----
    init_state_kernel<<<(BATCH_N * HID_N) / 256, 256, 0, stream>>>(
        h0f, h1f, h0_16[0], h1_16[0]);

    // ---- input projection: xp = x @ W_in^T + b_in  (M=16384,N=1024,K=512) ----
    {
        int mt = (SEQ_N * BATCH_N) / 16, ntN = HID_N / 16;      // 1024 x 64 tiles
        int blocks = (mt * (ntN / 2)) / 8;                      // 4096
        gemm_bias_kernel<0, true><<<blocks, 256, 0, stream>>>(
            x16, IN_N, Win16, IN_N, b_in, xp16, nullptr, HID_N, ntN, IN_N);
    }

    // ---- sequential scan ----
    const int liquid_blocks = (4 * ((HID_N / 16) / 2)) / 8;     // 16
    const int o1_blocks     = (4 * ((HID_N / 2 / 16) / 2)) / 8; // 8
    const int o2_blocks     = (4 * ((OUT_N / 16) / 2)) / 8;     // 4

    for (int t = 0; t < SEQ_N; ++t) {
        int cur = t & 1, nxt = (t + 1) & 1;
        const half_t* xpt = xp16 + (size_t)t * BATCH_N * HID_N;

        // layer 0: inp = xp_t
        liquid_step_kernel<<<liquid_blocks, 256, 0, stream>>>(
            xpt, h0_16[cur], h0f, h0_16[nxt], Wc16, b_c, taus + 0);
        // layer 1: inp = h0_new
        liquid_step_kernel<<<liquid_blocks, 256, 0, stream>>>(
            h0_16[nxt], h1_16[cur], h1f, h1_16[nxt], Wc16, b_c, taus + 1);
        // o1 = relu(h1_new @ W_o1^T + b_o1)   (M=64,N=512,K=1024)
        gemm_bias_kernel<1, true><<<o1_blocks, 256, 0, stream>>>(
            h1_16[nxt], HID_N, Wo116, HID_N, b_o1, o1_16, nullptr,
            HID_N / 2, (HID_N / 2) / 16, HID_N);
        // out_t = o1 @ W_o2^T + b_o2          (M=64,N=256,K=512) -> f32
        gemm_bias_kernel<0, false><<<o2_blocks, 256, 0, stream>>>(
            o1_16, HID_N / 2, Wo216, HID_N / 2, b_o2, nullptr,
            out + (size_t)t * BATCH_N * OUT_N, OUT_N, OUT_N / 16, HID_N / 2);
    }

    // ---- final hidden state [L,B,H] appended after outputs ----
    copy_hidden_kernel<<<(BATCH_N * HID_N) / 256, 256, 0, stream>>>(
        h0f, h1f, out + (size_t)SEQ_N * BATCH_N * OUT_N);
}
